// CausalSelfAttention_59914793779784
// MI455X (gfx1250) — matrix operand order repaired and checked by reading, not compiled, not run on hardware
//
#include <hip/hip_runtime.h>
#include <hip/hip_bf16.h>

// Problem constants (from reference): B=4, T=2048, E=1024, H=16, D=64
#define BB 4
#define TT 2048
#define EE 1024
#define HH 16
#define DD 64
#define NQKV (3 * EE)
#define MROWS (BB * TT) // 8192

typedef unsigned short u16;
typedef __attribute__((ext_vector_type(16))) __bf16 v16bf;
typedef __attribute__((ext_vector_type(8)))  float  v8f;

struct U32x8 { uint4 lo, hi; };

static __device__ __forceinline__ v16bf make_frag(uint4 lo, uint4 hi) {
  U32x8 t; t.lo = lo; t.hi = hi;
  return __builtin_bit_cast(v16bf, t);
}

static __device__ __forceinline__ v8f wmma_bf16(v16bf a, v16bf b, v8f c) {
  // (neg_a, A, neg_b, B, c_mod, C, reuse_a, reuse_b)
  return __builtin_amdgcn_wmma_f32_16x16x32_bf16(false, a, false, b, (short)0, c,
                                                 false, false);
}

static __device__ __forceinline__ u16 f2bf(float f) {
  unsigned u = __builtin_bit_cast(unsigned, f);
  unsigned r = u + 0x7FFFu + ((u >> 16) & 1u); // round-to-nearest-even
  return (u16)(r >> 16);
}

// A-fragment (16x32 bf16, M x K) from row-major [*, ld] bf16 matrix.
// ISA layout: lane holds row (lane&15); halves 0..7 = K kb..kb+7,
// halves 8..15 = K kb+16..kb+23, with kb = k0 + (lane>=16 ? 8 : 0).
static __device__ __forceinline__ v16bf load_a(const u16* base, int ld, int row0, int k0) {
  const int lane = threadIdx.x & 31;
  const u16* p = base + (size_t)(row0 + (lane & 15)) * ld + (k0 + ((lane >> 4) << 3));
  uint4 lo = *(const uint4*)p;
  uint4 hi = *(const uint4*)(p + 16);
  return make_frag(lo, hi);
}

// B-fragment (32x16 bf16, K x N) from row-major [*, ld] bf16 matrix.
// ISA layout: lane holds column (lane&15); halves 0..7 = K kb..kb+7,
// halves 8..15 = K kb+16..kb+23, with kb = k0 + (lane>=16 ? 8 : 0).
static __device__ __forceinline__ v16bf load_b(const u16* base, int ld, int k0, int n0) {
  const int lane = threadIdx.x & 31;
  const u16* p = base + (size_t)(k0 + ((lane >> 4) << 3)) * ld + n0 + (lane & 15);
  struct { u16 v[16]; } t;
#pragma unroll
  for (int s = 0; s < 8; ++s) {
    t.v[s] = p[(size_t)s * ld];
    t.v[8 + s] = p[(size_t)(16 + s) * ld];
  }
  return __builtin_bit_cast(v16bf, t);
}

// ---------------------------------------------------------------------------
// Stage 0: f32 -> bf16 conversion (x, W_attn, W_proj)
// ---------------------------------------------------------------------------
__global__ void cvt_f32_bf16(const float* __restrict__ src, u16* __restrict__ dst, int n) {
  int i = blockIdx.x * blockDim.x + threadIdx.x;
  const int stride = gridDim.x * blockDim.x;
  for (; i < n; i += stride) dst[i] = f2bf(src[i]);
}

// ---------------------------------------------------------------------------
// Stage 1: QKV GEMM (8192x1024 @ 1024x3072) + bias + RoPE + head scatter.
// Workgroup = 256 threads = 8 waves (2 M-waves x 4 N-waves), wave tile 32x64,
// workgroup tile 64x256.  Q -> [B,H,T,D], K -> [B,H,D,T] (transposed for
// B-frag-friendly QK^T), V -> [B,H,T,D], all bf16.
// ---------------------------------------------------------------------------
__global__ void __launch_bounds__(256)
qkv_rope_kernel(const u16* __restrict__ xb, const u16* __restrict__ Wa,
                const float* __restrict__ b_attn,
                u16* __restrict__ Q, u16* __restrict__ Kt, u16* __restrict__ V) {
  const int lane = threadIdx.x & 31;
  const int wid = threadIdx.x >> 5;
  const int waveM = wid >> 2, waveN = wid & 3;
  const int mBase = blockIdx.y * 64 + waveM * 32;
  const int nBase = blockIdx.x * 256 + waveN * 64;

  v8f zero = {0.f, 0.f, 0.f, 0.f, 0.f, 0.f, 0.f, 0.f};
  v8f acc[2][4];
#pragma unroll
  for (int mi = 0; mi < 2; ++mi)
#pragma unroll
    for (int ni = 0; ni < 4; ++ni) acc[mi][ni] = zero;

  for (int k0 = 0; k0 < EE; k0 += 32) {
    v16bf a0 = load_a(xb, EE, mBase, k0);
    v16bf a1 = load_a(xb, EE, mBase + 16, k0);
#pragma unroll
    for (int ni = 0; ni < 4; ++ni) {
      v16bf bf = load_b(Wa, NQKV, k0, nBase + ni * 16);
      acc[0][ni] = wmma_bf16(a0, bf, acc[0][ni]);
      acc[1][ni] = wmma_bf16(a1, bf, acc[1][ni]);
    }
  }

  const int rowoff = (lane >> 4) << 3;
#pragma unroll
  for (int mi = 0; mi < 2; ++mi) {
#pragma unroll
    for (int ni = 0; ni < 4; ++ni) {
      const int n = nBase + ni * 16 + (lane & 15); // uniform section per tile
      const float bias = b_attn[n];
      v8f a = acc[mi][ni];
#pragma unroll
      for (int r = 0; r < 8; ++r) {
        const int row = mBase + mi * 16 + rowoff + r;
        const int bidx = row >> 11;      // T = 2048
        const int t = row & (TT - 1);
        float val = a[r] + bias;
        if (n < 2 * EE) {
          // q or k -> RoPE; partner column (pair within head dim) is the
          // adjacent lane in the 16-lane N group.
          const float prt = __shfl_xor(val, 1);
          const int nn = (n < EE) ? n : (n - EE);
          const int dim = nn & (DD - 1);
          // inv_freq = 10000^(-(dim&~1)/D) = 2^(-(dim&~1)/D * log2(1e4))
          const float inv = exp2f((float)(dim & ~1) * (-13.28771237954945f / (float)DD));
          const float ang = (float)t * inv;
          float s, c;
          __sincosf(ang, &s, &c);
          val = val * c + ((dim & 1) ? prt * s : -prt * s);
          const int h = nn >> 6;
          if (n < EE) {
            Q[((size_t)((bidx * HH + h) * TT + t)) * DD + dim] = f2bf(val);
          } else {
            Kt[((size_t)((bidx * HH + h) * DD + dim)) * TT + t] = f2bf(val);
          }
        } else {
          const int nn = n - 2 * EE;
          const int h = nn >> 6, dim = nn & (DD - 1);
          V[((size_t)((bidx * HH + h) * TT + t)) * DD + dim] = f2bf(val);
        }
      }
    }
  }
}

// ---------------------------------------------------------------------------
// Stage 2: causal flash attention.  128 threads = 4 waves; each wave owns a
// 16-row Q tile and streams 32-key blocks with online softmax.  Per block:
// 4 QK^T WMMAs (D=64 contraction) + 4 P*V WMMAs.  P is moved from C-layout to
// A-layout via a per-wave LDS staging tile (no cross-wave sync needed).
// ---------------------------------------------------------------------------
#define FA_WAVES 4
__global__ void __launch_bounds__(128)
flash_attn_kernel(const u16* __restrict__ Q, const u16* __restrict__ Kt,
                  const u16* __restrict__ V, u16* __restrict__ Ctx) {
  __shared__ u16 pbuf[FA_WAVES * 16 * 32];
  const int lane = threadIdx.x & 31;
  const int wid = threadIdx.x >> 5;
  const int bh = blockIdx.y;           // b*H + h
  const int b = bh >> 4, h = bh & 15;  // H = 16
  const int q0 = (blockIdx.x * FA_WAVES + wid) * 16;

  const u16* Qp = Q + (size_t)bh * TT * DD;
  const u16* Kp = Kt + (size_t)bh * DD * TT;
  const u16* Vp = V + (size_t)bh * TT * DD;

  const v16bf qf0 = load_a(Qp, DD, q0, 0);
  const v16bf qf1 = load_a(Qp, DD, q0, 32);

  v8f zero = {0.f, 0.f, 0.f, 0.f, 0.f, 0.f, 0.f, 0.f};
  v8f o[4];
  float mrow[8], lrow[8];
#pragma unroll
  for (int dt = 0; dt < 4; ++dt) o[dt] = zero;
#pragma unroll
  for (int r = 0; r < 8; ++r) { mrow[r] = -3.0e38f; lrow[r] = 0.f; }

  const int rowoff = (lane >> 4) << 3;
  const int nblk = ((q0 + 15) >> 5) + 1; // causal: keys 0 .. q0+15

  for (int kb = 0; kb < nblk; ++kb) {
    const int c0 = kb * 32;
    // S = Q(16x64) @ K^T(64x32): two N-tiles of 16 keys, K-loop over d.
    v16bf kB00 = load_b(Kp, TT, 0,  c0);
    v16bf kB10 = load_b(Kp, TT, 32, c0);
    v16bf kB01 = load_b(Kp, TT, 0,  c0 + 16);
    v16bf kB11 = load_b(Kp, TT, 32, c0 + 16);
    v8f s0 = wmma_bf16(qf0, kB00, zero);
    s0 = wmma_bf16(qf1, kB10, s0);
    v8f s1 = wmma_bf16(qf0, kB01, zero);
    s1 = wmma_bf16(qf1, kB11, s1);

    const int cA = c0 + (lane & 15);
    const int cB = cA + 16;
    float alpha[8];
#pragma unroll
    for (int r = 0; r < 8; ++r) {
      const int trow = q0 + rowoff + r;
      float v0 = s0[r] * 0.125f; // 1/sqrt(64)
      float v1 = s1[r] * 0.125f;
      if (cA > trow) v0 = -3.0e38f;
      if (cB > trow) v1 = -3.0e38f;
      // Row max across the 16-lane N group (both N-tiles).
      float mloc = fmaxf(v0, v1);
#pragma unroll
      for (int off = 8; off; off >>= 1) mloc = fmaxf(mloc, __shfl_xor(mloc, off));
      const float mnew = fmaxf(mrow[r], mloc);
      const float al = __expf(mrow[r] - mnew);
      const float p0 = __expf(v0 - mnew);
      const float p1 = __expf(v1 - mnew);
      float ps = p0 + p1;
#pragma unroll
      for (int off = 8; off; off >>= 1) ps += __shfl_xor(ps, off);
      lrow[r] = lrow[r] * al + ps;
      mrow[r] = mnew;
      alpha[r] = al;
      // Stage P (bf16) into LDS in row-major 16x32 for A-frag pickup.
      const int prow = rowoff + r;
      pbuf[(wid * 16 + prow) * 32 + (lane & 15)] = f2bf(p0);
      pbuf[(wid * 16 + prow) * 32 + 16 + (lane & 15)] = f2bf(p1);
    }
    // Per-wave LDS RAW: DS ops from one wave are issued in order; the wait
    // covers VGPR availability of the re-load below.
    asm volatile("s_wait_dscnt 0" ::: "memory");
    const v16bf pf = load_a(&pbuf[wid * 16 * 32], 32, 0, 0);

    // Rescale running output by alpha, then accumulate P @ V.
#pragma unroll
    for (int dt = 0; dt < 4; ++dt) {
#pragma unroll
      for (int r = 0; r < 8; ++r) o[dt][r] *= alpha[r];
    }
#pragma unroll
    for (int dt = 0; dt < 4; ++dt) {
      v16bf vf = load_b(Vp, DD, c0, dt * 16);
      o[dt] = wmma_bf16(pf, vf, o[dt]);
    }
  }

  // Normalize and write context back as [B, T, E] bf16 (head-interleaved).
#pragma unroll
  for (int r = 0; r < 8; ++r) {
    const float inv = 1.0f / lrow[r];
    const int t = q0 + rowoff + r;
#pragma unroll
    for (int dt = 0; dt < 4; ++dt) {
      Ctx[((size_t)(b * TT + t)) * EE + h * DD + dt * 16 + (lane & 15)] =
          f2bf(o[dt][r] * inv);
    }
  }
}

// ---------------------------------------------------------------------------
// Stage 3: output projection (8192x1024 @ 1024x1024) + bias -> f32 out.
// ---------------------------------------------------------------------------
__global__ void __launch_bounds__(256)
proj_kernel(const u16* __restrict__ Cb, const u16* __restrict__ Wp,
            const float* __restrict__ b_proj, float* __restrict__ out) {
  const int lane = threadIdx.x & 31;
  const int wid = threadIdx.x >> 5;
  const int waveM = wid >> 2, waveN = wid & 3;
  const int mBase = blockIdx.y * 64 + waveM * 32;
  const int nBase = blockIdx.x * 256 + waveN * 64;

  v8f zero = {0.f, 0.f, 0.f, 0.f, 0.f, 0.f, 0.f, 0.f};
  v8f acc[2][4];
#pragma unroll
  for (int mi = 0; mi < 2; ++mi)
#pragma unroll
    for (int ni = 0; ni < 4; ++ni) acc[mi][ni] = zero;

  for (int k0 = 0; k0 < EE; k0 += 32) {
    v16bf a0 = load_a(Cb, EE, mBase, k0);
    v16bf a1 = load_a(Cb, EE, mBase + 16, k0);
#pragma unroll
    for (int ni = 0; ni < 4; ++ni) {
      v16bf bf = load_b(Wp, EE, k0, nBase + ni * 16);
      acc[0][ni] = wmma_bf16(a0, bf, acc[0][ni]);
      acc[1][ni] = wmma_bf16(a1, bf, acc[1][ni]);
    }
  }

  const int rowoff = (lane >> 4) << 3;
#pragma unroll
  for (int mi = 0; mi < 2; ++mi) {
#pragma unroll
    for (int ni = 0; ni < 4; ++ni) {
      const int n = nBase + ni * 16 + (lane & 15);
      const float bias = b_proj[n];
#pragma unroll
      for (int r = 0; r < 8; ++r) {
        const int row = mBase + mi * 16 + rowoff + r;
        out[(size_t)row * EE + n] = acc[mi][ni][r] + bias;
      }
    }
  }
}

// ---------------------------------------------------------------------------
extern "C" void kernel_launch(void* const* d_in, const int* in_sizes, int n_in,
                              void* d_out, int out_size, void* d_ws, size_t ws_size,
                              hipStream_t stream) {
  const float* x      = (const float*)d_in[0];
  const float* W_attn = (const float*)d_in[1];
  const float* b_attn = (const float*)d_in[2];
  const float* W_proj = (const float*)d_in[3];
  const float* b_proj = (const float*)d_in[4];
  float* out = (float*)d_out;

  // Workspace layout (bf16 halves).  Ctx aliases xb (x dead after QKV GEMM).
  u16* ws = (u16*)d_ws;
  u16* xb = ws;                                       // 8192*1024
  u16* Wa = xb + (size_t)MROWS * EE;                  // 1024*3072
  u16* Wp = Wa + (size_t)EE * NQKV;                   // 1024*1024
  u16* Qb = Wp + (size_t)EE * EE;                     // B*H*T*D
  u16* Kt = Qb + (size_t)BB * HH * TT * DD;           // B*H*D*T (transposed)
  u16* Vb = Kt + (size_t)BB * HH * TT * DD;           // B*H*T*D
  u16* Ctx = xb;                                      // reuse

  cvt_f32_bf16<<<dim3(2048), dim3(256), 0, stream>>>(x, xb, MROWS * EE);
  cvt_f32_bf16<<<dim3(1024), dim3(256), 0, stream>>>(W_attn, Wa, EE * NQKV);
  cvt_f32_bf16<<<dim3(512),  dim3(256), 0, stream>>>(W_proj, Wp, EE * EE);

  qkv_rope_kernel<<<dim3(NQKV / 256, MROWS / 64), dim3(256), 0, stream>>>(
      xb, Wa, b_attn, Qb, Kt, Vb);

  flash_attn_kernel<<<dim3(TT / (16 * FA_WAVES), BB * HH), dim3(128), 0, stream>>>(
      Qb, Kt, Vb, Ctx);

  proj_kernel<<<dim3(EE / 256, MROWS / 64), dim3(256), 0, stream>>>(
      Ctx, Wp, b_proj, out);
}
